// EncoderMoE_20418274525447
// MI455X (gfx1250) — matrix operand order repaired and checked by reading, not compiled, not run on hardware
//
#include <hip/hip_runtime.h>
#include <hip/hip_bf16.h>
#include <math.h>

// ---------------------------------------------------------------------------
// MI455X / gfx1250: wave32, WMMA 16x16x32 bf16 -> f32.
// Model: B=128 T=128 D=512 H=8 DH=64 NL=3 FF=2048 E=4 VOCAB=120 FEAT=200.
// All GEMM B-operands are weights, stored pre-transposed (WT[N][K]) so every
// WMMA fragment load is a pair of 16B ds_load_b128 ops.
// ---------------------------------------------------------------------------

typedef __attribute__((ext_vector_type(16))) __bf16 v16bf;
typedef __attribute__((ext_vector_type(8)))  float  v8f;

union BF16x16 { v16bf v; uint4 q[2]; };
union BF16x8  { uint4 q; __bf16 e[8]; };

#define BB     128
#define TT     128
#define DD     512
#define HH     8
#define DHH    64
#define FFD    2048
#define NE     4
#define NTOK   (BB * TT)   // 16384
#define FEAT   200
#define FEATP  224         // FEAT padded to 7*32 for K-loop

// GEMM epilogue flags (compile-time template parameter)
#define GF_BIAS     1
#define GF_RELU     2
#define GF_ACC      4
#define GF_ROWSCALE 8
#define GF_WRITEF   16
#define GF_WRITEB   32

// ---------------------------------------------------------------------------
// fp32 [K][N] -> bf16 transposed [N][K]
__global__ __launch_bounds__(256) void cvt_transpose(const float* __restrict__ in,
                                                     __bf16* __restrict__ out,
                                                     int K, int N) {
  int i = blockIdx.x * 256 + threadIdx.x;
  if (i >= K * N) return;
  int k = i / N, n = i % N;
  out[(size_t)n * K + k] = (__bf16)in[i];
}

// W_emb [FEAT,D] -> bf16 transposed+padded [D][FEATP]
__global__ __launch_bounds__(256) void pad_wemb_t(const float* __restrict__ in,
                                                  __bf16* __restrict__ out) {
  int i = blockIdx.x * 256 + threadIdx.x;
  if (i >= DD * FEATP) return;
  int n = i / FEATP, k = i % FEATP;
  out[i] = (k < FEAT) ? (__bf16)in[(size_t)k * DD + n] : (__bf16)0.0f;
}

// Gather cbfv[src] into padded bf16 A matrix [NTOK, FEATP]
__global__ __launch_bounds__(256) void gather_emb(const int* __restrict__ src,
                                                  const float* __restrict__ cbfv,
                                                  __bf16* __restrict__ Ae) {
  int i = blockIdx.x * 256 + threadIdx.x;
  if (i >= NTOK * FEATP) return;
  int t = i / FEATP, kk = i % FEATP;
  float v = (kk < FEAT) ? cbfv[(size_t)src[t] * FEAT + kk] : 0.0f;
  Ae[i] = (__bf16)v;
}

// ---------------------------------------------------------------------------
// bf16 WMMA GEMM: C[M,N] = op(A[M,K] @ WT[N,K]^T). Workgroup tile 128x128,
// 8 waves, wave tile 32x64 (2x4 WMMA 16x16x32). FLAGS is compile-time.
template <int FLAGS>
__global__ __launch_bounds__(256) void gemm_bf16(
    const __bf16* __restrict__ A, const __bf16* __restrict__ WT,
    const float* __restrict__ bias, const float* __restrict__ rowscale,
    int rsStride, float* __restrict__ C, __bf16* __restrict__ Cb,
    int M, int N, int K) {
  __shared__ __bf16 As[128][56];    // [m][k], 112B row stride (16B aligned)
  __shared__ __bf16 Bt[128][40];    // [n][k],  80B row stride (16B aligned)

  const int tid = threadIdx.x;
  const int wid = tid >> 5, lane = tid & 31;
  const int half = lane >> 4, l16 = lane & 15;
  const int m0 = blockIdx.y * 128, n0 = blockIdx.x * 128;
  const int wm = (wid >> 1) * 32, wn = (wid & 1) * 64;

  v8f zero = {};
  v8f acc[2][4];
  for (int mt = 0; mt < 2; ++mt)
    for (int nt = 0; nt < 4; ++nt) acc[mt][nt] = zero;

  const int r = tid >> 1, c0 = (tid & 1) * 16;  // staging assignment

  for (int kt = 0; kt < K; kt += 32) {
    { // stage A tile: 16 contiguous bf16 per thread, 2x b128 each way
      const uint4* sA = (const uint4*)(A + (size_t)(m0 + r) * K + kt + c0);
      *(uint4*)&As[r][c0]     = sA[0];
      *(uint4*)&As[r][c0 + 8] = sA[1];
    }
    { // stage B tile from transposed weights (row n, contiguous k)
      const uint4* sB = (const uint4*)(WT + (size_t)(n0 + r) * K + kt + c0);
      *(uint4*)&Bt[r][c0]     = sB[0];
      *(uint4*)&Bt[r][c0 + 8] = sB[1];
    }
    if (kt + 32 < K) { // prefetch next K tile (global_prefetch_b8)
      __builtin_prefetch(A + (size_t)(m0 + r) * K + kt + 32 + c0, 0, 1);
      __builtin_prefetch(WT + (size_t)(n0 + r) * K + kt + 32 + c0, 0, 1);
    }
    __syncthreads();

    BF16x16 af[2], bf[4];
#pragma unroll
    for (int mt = 0; mt < 2; ++mt) {   // A 16x32 frag: K split by lane half
      const __bf16* p = &As[wm + mt * 16 + l16][0];
      af[mt].q[0] = *(const uint4*)(p + half * 16);
      af[mt].q[1] = *(const uint4*)(p + half * 16 + 8);
    }
#pragma unroll
    for (int nt = 0; nt < 4; ++nt) {   // B 32x16 frag: contiguous K run
      const __bf16* p = &Bt[wn + nt * 16 + l16][half * 16];
      bf[nt].q[0] = *(const uint4*)p;
      bf[nt].q[1] = *(const uint4*)(p + 8);
    }
#pragma unroll
    for (int mt = 0; mt < 2; ++mt)
#pragma unroll
      for (int nt = 0; nt < 4; ++nt)
        acc[mt][nt] = __builtin_amdgcn_wmma_f32_16x16x32_bf16(
            false, af[mt].v, false, bf[nt].v, (short)0, acc[mt][nt], false, false);
    __syncthreads();
  }

  // Epilogue. C layout: VGPR v -> row v + 8*half, col = lane&15.
#pragma unroll
  for (int nt = 0; nt < 4; ++nt) {
    const int gc = n0 + wn + nt * 16 + l16;
    float bv = 0.0f;
    if constexpr (FLAGS & GF_BIAS) bv = bias[gc];
#pragma unroll
    for (int mt = 0; mt < 2; ++mt)
#pragma unroll
      for (int v = 0; v < 8; ++v) {
        const int gr = m0 + wm + mt * 16 + v + half * 8;
        float val = acc[mt][nt][v] + bv;
        if constexpr (FLAGS & GF_RELU)     val = fmaxf(val, 0.0f);
        if constexpr (FLAGS & GF_ROWSCALE) val *= rowscale[(size_t)gr * rsStride];
        const size_t off = (size_t)gr * N + gc;
        if constexpr (FLAGS & GF_ACC)         C[off] += val;
        else if constexpr (FLAGS & GF_WRITEF) C[off] = val;
        if constexpr (FLAGS & GF_WRITEB)      Cb[off] = (__bf16)val;
      }
  }
}

// ---------------------------------------------------------------------------
// Fused attention: one block per (batch, head). 8 waves x 16 query rows.
__global__ __launch_bounds__(256) void attention_kernel(
    const __bf16* __restrict__ qb, const __bf16* __restrict__ kb,
    const __bf16* __restrict__ vb, const int* __restrict__ src,
    const float* __restrict__ frac, const float* __restrict__ alphaP,
    __bf16* __restrict__ attnb) {
  __shared__ float  fracS[TT];
  __shared__ float  maskS[TT];
  __shared__ __bf16 Vt[DHH][TT + 8];     // V transposed [dh][token]
  __shared__ __bf16 Ps[8][16][TT + 8];   // per-wave softmax tile [16][128]

  const int bh = blockIdx.x;
  const int bi = bh >> 3, h = bh & 7;
  const int tid = threadIdx.x;
  const int wid = tid >> 5, lane = tid & 31;
  const int half = lane >> 4, l16 = lane & 15;
  const float alpha = alphaP[0];
  const float scale = 0.125f;            // DH^-0.5

  if (tid < TT) {
    fracS[tid] = frac[bi * TT + tid];
    maskS[tid] = (src[bi * TT + tid] == 0) ? -1e30f : 0.0f;
  }
  // stage V transposed: vector global load, scalar LDS transpose stores
  for (int i = tid; i < TT * (DHH / 8); i += 256) {
    int tok = i >> 3, d0 = (i & 7) * 8;
    BF16x8 t;
    t.q = *(const uint4*)(vb + ((size_t)(bi * TT + tok) * HH + h) * DHH + d0);
#pragma unroll
    for (int j = 0; j < 8; ++j) Vt[d0 + j][tok] = t.e[j];
  }
  __syncthreads();

  const int qr0 = wid * 16;

  // ---- scores = Q @ K^T : M=16 (per wave), N=128, K=64 ----
  v8f zero = {};
  v8f acc[8];
#pragma unroll
  for (int nt = 0; nt < 8; ++nt) acc[nt] = zero;

#pragma unroll
  for (int ks = 0; ks < 2; ++ks) {
    BF16x16 af;
    const __bf16* qrow =
        qb + ((size_t)(bi * TT + qr0 + l16) * HH + h) * DHH + ks * 32;
    af.q[0] = *(const uint4*)(qrow + half * 16);
    af.q[1] = *(const uint4*)(qrow + half * 16 + 8);
#pragma unroll
    for (int nt = 0; nt < 8; ++nt) {
      BF16x16 bf;
      const __bf16* krow =
          kb + ((size_t)(bi * TT + nt * 16 + l16) * HH + h) * DHH + ks * 32 + half * 16;
      bf.q[0] = *(const uint4*)krow;
      bf.q[1] = *(const uint4*)(krow + 8);
      acc[nt] = __builtin_amdgcn_wmma_f32_16x16x32_bf16(
          false, af.v, false, bf.v, (short)0, acc[nt], false, false);
    }
  }

  // ---- bias + mask + row softmax (row at (v,half); col at lane&15) ----
  float sc[8][8];
#pragma unroll
  for (int v = 0; v < 8; ++v) {
    float fq = fracS[qr0 + v + half * 8];
#pragma unroll
    for (int nt = 0; nt < 8; ++nt) {
      int col = nt * 16 + l16;
      float d = fq - fracS[col];
      float sg = (d > 0.0f) ? 1.0f : ((d < 0.0f) ? -1.0f : 0.0f);
      sc[nt][v] = acc[nt][v] * scale + alpha * log1pf(fabsf(d)) * sg + maskS[col];
    }
  }
#pragma unroll
  for (int v = 0; v < 8; ++v) {
    float m = -1e30f;
#pragma unroll
    for (int nt = 0; nt < 8; ++nt) m = fmaxf(m, sc[nt][v]);
    for (int off = 8; off >= 1; off >>= 1) m = fmaxf(m, __shfl_xor(m, off, 32));
    float sum = 0.0f;
#pragma unroll
    for (int nt = 0; nt < 8; ++nt) {
      float p = __expf(sc[nt][v] - m);
      sc[nt][v] = p;
      sum += p;
    }
    for (int off = 8; off >= 1; off >>= 1) sum += __shfl_xor(sum, off, 32);
    float inv = 1.0f / sum;
#pragma unroll
    for (int nt = 0; nt < 8; ++nt)
      Ps[wid][v + half * 8][nt * 16 + l16] = (__bf16)(sc[nt][v] * inv);
  }
  __syncthreads();

  // ---- out = P @ V : M=16, N=64, K=128 ----
  v8f oacc[4];
#pragma unroll
  for (int nt = 0; nt < 4; ++nt) oacc[nt] = zero;
#pragma unroll
  for (int ks = 0; ks < 4; ++ks) {
    BF16x16 af;
    const __bf16* prow = &Ps[wid][l16][ks * 32];
    af.q[0] = *(const uint4*)(prow + half * 16);
    af.q[1] = *(const uint4*)(prow + half * 16 + 8);
#pragma unroll
    for (int nt = 0; nt < 4; ++nt) {
      BF16x16 bf;
      const __bf16* vrow = &Vt[nt * 16 + l16][ks * 32 + half * 16];
      bf.q[0] = *(const uint4*)vrow;
      bf.q[1] = *(const uint4*)(vrow + 8);
      oacc[nt] = __builtin_amdgcn_wmma_f32_16x16x32_bf16(
          false, af.v, false, bf.v, (short)0, oacc[nt], false, false);
    }
  }
#pragma unroll
  for (int nt = 0; nt < 4; ++nt)
#pragma unroll
    for (int v = 0; v < 8; ++v) {
      int q = qr0 + v + half * 8;
      attnb[(size_t)(bi * TT + q) * DD + h * DHH + nt * 16 + l16] =
          (__bf16)oacc[nt][v];
    }
}

// ---------------------------------------------------------------------------
// x = LayerNorm(x + delta) ; one block per token, 256 threads (2 elems each)
__global__ __launch_bounds__(256) void add_ln(
    const float* __restrict__ xin, const float* __restrict__ delta,
    const float* __restrict__ g, const float* __restrict__ bb,
    float* __restrict__ xout, __bf16* __restrict__ xbout) {
  __shared__ float red[8];
  const int row = blockIdx.x;
  const int tid = threadIdx.x, wid = tid >> 5, lane = tid & 31;
  const size_t base = (size_t)row * DD;
  float v0 = xin[base + tid] + delta[base + tid];
  float v1 = xin[base + 256 + tid] + delta[base + 256 + tid];
  float s = v0 + v1;
  for (int off = 16; off; off >>= 1) s += __shfl_xor(s, off, 32);
  if (lane == 0) red[wid] = s;
  __syncthreads();
  float tot = 0.0f;
  for (int i = 0; i < 8; ++i) tot += red[i];
  float mean = tot * (1.0f / 512.0f);
  __syncthreads();
  float d0 = v0 - mean, d1 = v1 - mean;
  s = d0 * d0 + d1 * d1;
  for (int off = 16; off; off >>= 1) s += __shfl_xor(s, off, 32);
  if (lane == 0) red[wid] = s;
  __syncthreads();
  tot = 0.0f;
  for (int i = 0; i < 8; ++i) tot += red[i];
  float inv = rsqrtf(tot * (1.0f / 512.0f) + 1e-5f);
  float o0 = d0 * inv * g[tid] + bb[tid];
  float o1 = d1 * inv * g[256 + tid] + bb[256 + tid];
  xout[base + tid] = o0;
  xout[base + 256 + tid] = o1;
  xbout[base + tid] = (__bf16)o0;
  xbout[base + 256 + tid] = (__bf16)o1;
}

// ---------------------------------------------------------------------------
// Top-2 MoE gate weights: w[t,e] = softmax(x@Wg+bg)[e] if e in top2 else 0
__global__ __launch_bounds__(256) void gate_topk(
    const float* __restrict__ x, const float* __restrict__ Wg,
    const float* __restrict__ bg, float* __restrict__ w) {
  int t = blockIdx.x * 256 + threadIdx.x;
  if (t >= NTOK) return;
  float gsum[NE];
  for (int e = 0; e < NE; ++e) gsum[e] = bg[e];
  const float* xr = x + (size_t)t * DD;
  for (int i = 0; i < DD; ++i) {
    float xv = xr[i];
    for (int e = 0; e < NE; ++e) gsum[e] += xv * Wg[i * NE + e];
  }
  float m = gsum[0];
  for (int e = 1; e < NE; ++e) m = fmaxf(m, gsum[e]);
  float ssum = 0.0f;
  for (int e = 0; e < NE; ++e) { gsum[e] = __expf(gsum[e] - m); ssum += gsum[e]; }
  float inv = 1.0f / ssum;
  for (int e = 0; e < NE; ++e) gsum[e] *= inv;
  int i1 = 0;
  for (int e = 1; e < NE; ++e) if (gsum[e] > gsum[i1]) i1 = e;
  int i2 = -1;
  for (int e = 0; e < NE; ++e)
    if (e != i1 && (i2 < 0 || gsum[e] > gsum[i2])) i2 = e;
  for (int e = 0; e < NE; ++e)
    w[(size_t)t * NE + e] = (e == i1 || e == i2) ? gsum[e] : 0.0f;
}

// ---------------------------------------------------------------------------
template <int FLAGS>
static void launch_gemm(const __bf16* A, const __bf16* WT, const float* bias,
                        const float* rs, int rsStride, float* C, __bf16* Cb,
                        int M, int N, int K, hipStream_t stream) {
  dim3 grid(N / 128, M / 128);
  gemm_bf16<FLAGS><<<grid, dim3(256), 0, stream>>>(A, WT, bias, rs, rsStride,
                                                   C, Cb, M, N, K);
}

extern "C" void kernel_launch(void* const* d_in, const int* in_sizes, int n_in,
                              void* d_out, int out_size, void* d_ws, size_t ws_size,
                              hipStream_t stream) {
  (void)in_sizes; (void)n_in; (void)out_size; (void)ws_size;
  const int*   src    = (const int*)  d_in[0];
  const float* frac   = (const float*)d_in[1];
  const float* cbfv   = (const float*)d_in[2];
  const float* W_emb  = (const float*)d_in[3];
  const float* b_emb  = (const float*)d_in[4];
  const float* Wq     = (const float*)d_in[5];
  const float* bq     = (const float*)d_in[6];
  const float* Wk     = (const float*)d_in[7];
  const float* bk     = (const float*)d_in[8];
  const float* Wv     = (const float*)d_in[9];
  const float* bv     = (const float*)d_in[10];
  const float* Wo     = (const float*)d_in[11];
  const float* bo     = (const float*)d_in[12];
  const float* alphaP = (const float*)d_in[13];
  const float* ln1_g  = (const float*)d_in[14];
  const float* ln1_b  = (const float*)d_in[15];
  const float* ln2_g  = (const float*)d_in[16];
  const float* ln2_b  = (const float*)d_in[17];
  const float* Wg     = (const float*)d_in[18];
  const float* bg     = (const float*)d_in[19];
  const float* W1     = (const float*)d_in[20];
  const float* b1     = (const float*)d_in[21];
  const float* W2     = (const float*)d_in[22];
  const float* b2     = (const float*)d_in[23];
  float* out = (float*)d_out;

  // Workspace carve-up (~178 MB). h aliases [qb..attnb] (exactly 64 MB),
  // which are dead by the time MoE runs each layer.
  char* ws = (char*)d_ws;
  auto carve = [&](size_t bytes) {
    char* p = ws;
    ws += (bytes + 255) & ~(size_t)255;
    return p;
  };
  __bf16* xb    = (__bf16*)carve((size_t)NTOK * DD * 2);
  __bf16* qb    = (__bf16*)carve((size_t)NTOK * DD * 2);
  __bf16* kb    = (__bf16*)carve((size_t)NTOK * DD * 2);
  __bf16* vb    = (__bf16*)carve((size_t)NTOK * DD * 2);
  __bf16* attnb = (__bf16*)carve((size_t)NTOK * DD * 2);
  __bf16* hb    = qb;  // [NTOK, FFD] bf16 == 64 MB alias of qb..attnb
  float*  x     = (float*)carve((size_t)NTOK * DD * 4);
  float*  delta = (float*)carve((size_t)NTOK * DD * 4);
  __bf16* Ae    = (__bf16*)carve((size_t)NTOK * FEATP * 2);
  __bf16* WembT = (__bf16*)carve((size_t)DD * FEATP * 2);
  __bf16* WqT   = (__bf16*)carve((size_t)DD * DD * 2);
  __bf16* WkT   = (__bf16*)carve((size_t)DD * DD * 2);
  __bf16* WvT   = (__bf16*)carve((size_t)DD * DD * 2);
  __bf16* WoT   = (__bf16*)carve((size_t)DD * DD * 2);
  __bf16* W1T   = (__bf16*)carve((size_t)NE * DD * FFD * 2);   // [e][FF][D]
  __bf16* W2T   = (__bf16*)carve((size_t)NE * FFD * DD * 2);   // [e][D][FF]
  float*  wgate = (float*)carve((size_t)NTOK * NE * 4);

  auto cvtT = [&](const float* in, __bf16* o, int K, int N) {
    cvt_transpose<<<(K * N + 255) / 256, 256, 0, stream>>>(in, o, K, N);
  };
  cvtT(Wq, WqT, DD, DD);
  cvtT(Wk, WkT, DD, DD);
  cvtT(Wv, WvT, DD, DD);
  cvtT(Wo, WoT, DD, DD);
  for (int e = 0; e < NE; ++e) {
    cvtT(W1 + (size_t)e * DD * FFD, W1T + (size_t)e * FFD * DD, DD, FFD);
    cvtT(W2 + (size_t)e * FFD * DD, W2T + (size_t)e * DD * FFD, FFD, DD);
  }
  pad_wemb_t<<<(DD * FEATP + 255) / 256, 256, 0, stream>>>(W_emb, WembT);
  gather_emb<<<(NTOK * FEATP + 255) / 256, 256, 0, stream>>>(src, cbfv, Ae);

  dim3 blk(256);

  // Embedder: x = cbfv[src] @ W_emb + b_emb
  launch_gemm<GF_BIAS | GF_WRITEF | GF_WRITEB>(
      Ae, WembT, b_emb, nullptr, 0, x, xb, NTOK, DD, FEATP, stream);

  for (int layer = 0; layer < 3; ++layer) {
    // Q,K,V projections (bf16 outputs feed attention WMMA)
    launch_gemm<GF_BIAS | GF_WRITEB>(xb, WqT, bq, nullptr, 0, nullptr, qb,
                                     NTOK, DD, DD, stream);
    launch_gemm<GF_BIAS | GF_WRITEB>(xb, WkT, bk, nullptr, 0, nullptr, kb,
                                     NTOK, DD, DD, stream);
    launch_gemm<GF_BIAS | GF_WRITEB>(xb, WvT, bv, nullptr, 0, nullptr, vb,
                                     NTOK, DD, DD, stream);

    attention_kernel<<<BB * HH, blk, 0, stream>>>(qb, kb, vb, src, frac, alphaP,
                                                  attnb);

    // Output projection, then residual + LN1
    launch_gemm<GF_BIAS | GF_WRITEF>(attnb, WoT, bo, nullptr, 0, delta, nullptr,
                                     NTOK, DD, DD, stream);
    add_ln<<<NTOK, blk, 0, stream>>>(x, delta, ln1_g, ln1_b, x, xb);

    // Top-2 gate weights
    gate_topk<<<NTOK / 256, blk, 0, stream>>>(x, Wg, bg, wgate);

    // Dense top-2 MoE: delta = sum_e w[:,e] * (relu(x@W1[e]+b1[e])@W2[e]+b2[e])
    for (int e = 0; e < NE; ++e) {
      launch_gemm<GF_BIAS | GF_RELU | GF_WRITEB>(
          xb, W1T + (size_t)e * FFD * DD, b1 + (size_t)e * FFD, nullptr, 0,
          nullptr, hb, NTOK, FFD, DD, stream);
      if (e == 0)
        launch_gemm<GF_BIAS | GF_ROWSCALE | GF_WRITEF>(
            hb, W2T + (size_t)e * DD * FFD, b2 + (size_t)e * DD, wgate + e, NE,
            delta, nullptr, NTOK, DD, FFD, stream);
      else
        launch_gemm<GF_BIAS | GF_ROWSCALE | GF_ACC>(
            hb, W2T + (size_t)e * DD * FFD, b2 + (size_t)e * DD, wgate + e, NE,
            delta, nullptr, NTOK, DD, FFD, stream);
    }

    // Residual + LN2 (final layer writes d_out directly)
    float* xdst = (layer == 2) ? out : x;
    add_ln<<<NTOK, blk, 0, stream>>>(x, delta, ln2_g, ln2_b, xdst, xb);
  }
}
